// LinkPrediction_65730179498528
// MI455X (gfx1250) — matrix operand order repaired and checked by reading, not compiled, not run on hardware
//
#include <hip/hip_runtime.h>
#include <hip/hip_bf16.h>

typedef __attribute__((ext_vector_type(16))) __bf16 v16bf;
typedef __attribute__((ext_vector_type(8)))  float  v8f;

#define DIM 128
#define LDS_STRIDE 136   // 128 + 8 bf16 pad: 272B row stride

// ---------------- elementwise / data-movement kernels ----------------

__global__ void k_zero(float* __restrict__ p, long n) {
    long i = ((long)blockIdx.x * blockDim.x + threadIdx.x) * 4;
    if (i + 3 < n) {
        *(float4*)(p + i) = make_float4(0.f, 0.f, 0.f, 0.f);
    } else {
        for (long j = i; j < n; ++j) p[j] = 0.f;
    }
}

// out[row][:] = emb[ids[row]][:]   (one wave of 32 lanes per 128-float row)
__global__ void k_gather(const float* __restrict__ emb, const int* __restrict__ ids,
                         float* __restrict__ out, int n) {
    long t = (long)blockIdx.x * blockDim.x + threadIdx.x;
    int row = (int)(t >> 5);
    int lane = (int)(t & 31);
    if (row >= n) return;
    long src = ids[row];
    float4 v = *(const float4*)(emb + src * DIM + lane * 4);
    *(float4*)(out + (long)row * DIM + lane * 4) = v;
}

// counts per destination node
__global__ void k_count(const int* __restrict__ dst, float* __restrict__ cnt, int E) {
    int e = blockIdx.x * blockDim.x + threadIdx.x;
    if (e >= E) return;
    atomicAdd(cnt + dst[e], 1.0f);
}

__global__ void k_inv(float* __restrict__ cnt, int n) {
    int i = blockIdx.x * blockDim.x + threadIdx.x;
    if (i >= n) return;
    cnt[i] = 1.0f / fmaxf(cnt[i], 1.0f);
}

// agg[dst[e]][:] += x[src[e]][:]   (wave per edge, float4 atomics)
__global__ void k_scatter(const float* __restrict__ x, const int* __restrict__ src,
                          const int* __restrict__ dst, float* __restrict__ agg, int E) {
    long t = (long)blockIdx.x * blockDim.x + threadIdx.x;
    int e = (int)(t >> 5);
    int lane = (int)(t & 31);
    if (e >= E) return;
    long s = src[e];
    long d = dst[e];
    float4 v = *(const float4*)(x + s * DIM + lane * 4);
    float* o = agg + d * DIM + lane * 4;
    atomicAdd(o + 0, v.x);
    atomicAdd(o + 1, v.y);
    atomicAdd(o + 2, v.z);
    atomicAdd(o + 3, v.w);
}

// agg[row][:] *= inv[row]
__global__ void k_scale(float* __restrict__ agg, const float* __restrict__ inv, int n) {
    long t = (long)blockIdx.x * blockDim.x + threadIdx.x;
    int row = (int)(t >> 5);
    int lane = (int)(t & 31);
    if (row >= n) return;
    float s = inv[row];
    float4* p = (float4*)(agg + (long)row * DIM + lane * 4);
    float4 v = *p;
    v.x *= s; v.y *= s; v.z *= s; v.w *= s;
    *p = v;
}

// transpose + convert weights into a stacked panel:
// W[K x 128] f32 -> Wt[n][koff + k] bf16, output row stride Ktot (N=128 rows)
__global__ void k_prep_w(const float* __restrict__ W, __bf16* __restrict__ Wt,
                         int K, int Ktot, int koff) {
    int t = blockIdx.x * blockDim.x + threadIdx.x;
    if (t >= K * DIM) return;
    int k = t >> 7;       // input-dim index
    int n = t & 127;      // output-dim index
    Wt[(long)n * Ktot + koff + k] = (__bf16)W[(long)k * DIM + n];
}

// ---------------- WMMA GEMM: C[M x 128] (+)= [A0 | A1] @ W + b1 + b2 ----------------
// Split-A: A0 is M x K0, A1 is M x K1 (K1 may be 0). Wt is 128 x (K0+K1) bf16.
// Weight panel staged in LDS in 128-wide K chunks; all 4 A fragments of a chunk are
// preloaded, and the B fragment is double-buffered across the 32-WMMA chain so DS
// latency overlaps WMMA issue (wait dscnt<=2 instead of 0).
// flags bit0: accumulate into existing C; bit1: relu epilogue.
// Block = 256 threads = 8 waves; each wave owns a 16-row strip, block covers 128 rows.
__global__ __launch_bounds__(256)
void k_gemm_bf16(const float* __restrict__ A0, int K0,
                 const float* __restrict__ A1, int K1,
                 const __bf16* __restrict__ Wt,
                 const float* __restrict__ b1, const float* __restrict__ b2,
                 float* __restrict__ C, int M, int flags) {
    extern __shared__ __bf16 bsh[];             // 128 rows x LDS_STRIDE bf16

    const int K = K0 + K1;
    const int tid  = threadIdx.x;
    const int wave = tid >> 5;
    const int lane = tid & 31;
    const int lhalf = lane >> 4;                // 0 or 1
    const int lmod  = lane & 15;

    const int tileRow = (blockIdx.x * 8 + wave) * 16;
    const bool active = (tileRow < M);          // wave-uniform predicate

    int rowA = tileRow + lmod;
    if (rowA >= M) rowA = M - 1;                // clamp; guarded at store

    v8f acc[8];
#pragma unroll
    for (int nt = 0; nt < 8; ++nt)
#pragma unroll
        for (int v = 0; v < 8; ++v) acc[nt][v] = 0.f;

    union BU { uint4 q[2]; v16bf v; };
    const __bf16* __restrict__ bcol = bsh + lmod * LDS_STRIDE + lhalf * 8;

    for (int k0 = 0; k0 < K; k0 += 128) {
        // ---- cooperative stage of Wt[:, k0..k0+127] into padded LDS ----
        for (int t = tid; t < 128 * 16; t += 256) {      // 16 x 16B chunks per row
            int row = t >> 4;
            int c   = t & 15;
            *(uint4*)(bsh + row * LDS_STRIDE + c * 8) =
                *(const uint4*)(Wt + (long)row * K + k0 + c * 8);
        }
        __syncthreads();

        if (active) {
            // pick which A matrix this 128-wide K slice comes from
            const float* __restrict__ abase =
                (k0 < K0) ? (A0 + (long)rowA * K0 + k0)
                          : (A1 + (long)rowA * K1 + (k0 - K0));

            // preload all 4 A fragments of this chunk (one load clause)
            v16bf a[4];
#pragma unroll
            for (int kq = 0; kq < 4; ++kq) {
                const float* ap = abase + kq * 32 + lhalf * 8;
                float4 x0 = *(const float4*)(ap);
                float4 x1 = *(const float4*)(ap + 4);
                float4 y0 = *(const float4*)(ap + 16);
                float4 y1 = *(const float4*)(ap + 20);
                a[kq][0] = (__bf16)x0.x; a[kq][1] = (__bf16)x0.y;
                a[kq][2] = (__bf16)x0.z; a[kq][3] = (__bf16)x0.w;
                a[kq][4] = (__bf16)x1.x; a[kq][5] = (__bf16)x1.y;
                a[kq][6] = (__bf16)x1.z; a[kq][7] = (__bf16)x1.w;
                a[kq][8]  = (__bf16)y0.x; a[kq][9]  = (__bf16)y0.y;
                a[kq][10] = (__bf16)y0.z; a[kq][11] = (__bf16)y0.w;
                a[kq][12] = (__bf16)y1.x; a[kq][13] = (__bf16)y1.y;
                a[kq][14] = (__bf16)y1.z; a[kq][15] = (__bf16)y1.w;
            }

            // 32-step WMMA chain (kq major, nt minor) with double-buffered B
            BU b[2];
            {
                const __bf16* bp = bcol;             // kq=0, nt=0
                b[0].q[0] = *(const uint4*)bp;
                b[0].q[1] = *(const uint4*)(bp + 16);
            }
#pragma unroll
            for (int idx = 0; idx < 32; ++idx) {
                const int kq = idx >> 3;
                const int nt = idx & 7;
                if (idx + 1 < 32) {
                    const int kq1 = (idx + 1) >> 3;
                    const int nt1 = (idx + 1) & 7;
                    const __bf16* bp = bcol + nt1 * (16 * LDS_STRIDE) + kq1 * 32;
                    b[(idx + 1) & 1].q[0] = *(const uint4*)bp;
                    b[(idx + 1) & 1].q[1] = *(const uint4*)(bp + 16);
                }
                acc[nt] = __builtin_amdgcn_wmma_f32_16x16x32_bf16(
                    false, a[kq], false, b[idx & 1].v, (short)0, acc[nt], false, false);
            }
        }
        __syncthreads();
    }

    if (active) {
        // epilogue: C layout — VGPR v, lane l -> row v + 8*(l>>4), col l&15
#pragma unroll
        for (int nt = 0; nt < 8; ++nt) {
            const int col = nt * 16 + lmod;
            float bv = b1 ? b1[col] : 0.f;
            if (b2) bv += b2[col];
#pragma unroll
            for (int v = 0; v < 8; ++v) {
                const int row = tileRow + v + lhalf * 8;
                if (row < M) {
                    float r = acc[nt][v] + bv;
                    long idx = (long)row * DIM + col;
                    if (flags & 1) r += C[idx];
                    if (flags & 2) r = fmaxf(r, 0.f);
                    C[idx] = r;
                }
            }
        }
    }
}

// pred[e] = dot(ha[ia[e]], hp[ip[e]])   (wave per edge)
__global__ void k_dot(const float* __restrict__ ha, const float* __restrict__ hp,
                      const int* __restrict__ ia, const int* __restrict__ ip,
                      float* __restrict__ out, int E) {
    long t = (long)blockIdx.x * blockDim.x + threadIdx.x;
    int e = (int)(t >> 5);
    int lane = (int)(t & 31);
    if (e >= E) return;
    float4 a = *(const float4*)(ha + (long)ia[e] * DIM + lane * 4);
    float4 p = *(const float4*)(hp + (long)ip[e] * DIM + lane * 4);
    float s = a.x * p.x + a.y * p.y + a.z * p.z + a.w * p.w;
#pragma unroll
    for (int off = 16; off > 0; off >>= 1) s += __shfl_xor(s, off, 32);
    if (lane == 0) out[e] = s;
}

// ---------------- host orchestration ----------------

static inline int cdiv(long a, long b) { return (int)((a + b - 1) / b); }

extern "C" void kernel_launch(void* const* d_in, const int* in_sizes, int n_in,
                              void* d_out, int out_size, void* d_ws, size_t ws_size,
                              hipStream_t stream) {
    const float* paper_x    = (const float*)d_in[0];
    const int*   author_nid = (const int*)d_in[1];
    const int*   paper_nid  = (const int*)d_in[2];
    const int*   domain_nid = (const int*)d_in[3];
    const int*   inst_nid   = (const int*)d_in[4];
    const int*   ew         = (const int*)d_in[5];
    const int*   ea         = (const int*)d_in[6];
    const int*   et         = (const int*)d_in[7];
    const int*   eli        = (const int*)d_in[8];
    const float* author_emb = (const float*)d_in[9];
    const float* paper_emb  = (const float*)d_in[10];
    const float* domain_emb = (const float*)d_in[11];
    const float* inst_emb   = (const float*)d_in[12];
    const float* lin_W      = (const float*)d_in[13];
    const float* lin_b      = (const float*)d_in[14];
    const float* Wl1 = (const float*)d_in[15]; const float* bl1 = (const float*)d_in[16];
    const float* Wr1 = (const float*)d_in[17]; const float* br1 = (const float*)d_in[18];
    const float* Wl2 = (const float*)d_in[19]; const float* bl2 = (const float*)d_in[20];
    const float* Wr2 = (const float*)d_in[21]; const float* br2 = (const float*)d_in[22];

    const int NA = in_sizes[1];
    const int NP = in_sizes[2];
    const int ND = in_sizes[3];
    const int NI = in_sizes[4];
    const int EW = in_sizes[5] / 2;
    const int EA = in_sizes[6] / 2;
    const int ET = in_sizes[7] / 2;
    const int EL = in_sizes[8] / 2;
    const int F  = in_sizes[0] / NP;

    // ---- workspace carve-up ----
    char* ws = (char*)d_ws;
    size_t off = 0;
    auto take = [&](size_t bytes) -> char* {
        char* p = ws + off;
        off += (bytes + 255) & ~(size_t)255;
        return p;
    };
    float* Xa = (float*)take((size_t)NA * DIM * 4);
    float* Xp = (float*)take((size_t)NP * DIM * 4);
    float* Xi = (float*)take((size_t)NI * DIM * 4);
    float* Xd = (float*)take((size_t)ND * DIM * 4);
    float* Ha = (float*)take((size_t)NA * DIM * 4);
    float* Hp = (float*)take((size_t)NP * DIM * 4);
    float* Hi = (float*)take((size_t)NI * DIM * 4);
    float* Hd = (float*)take((size_t)ND * DIM * 4);
    long maxN = NA > NP ? NA : NP;
    float* AGG = (float*)take((size_t)maxN * DIM * 4);

    // edge directions: src row / dst row, counts, types (0=author,1=paper,2=inst,3=domain)
    struct Dir { const int* s; const int* d; int E; int st; int dt; };
    Dir dirs[6] = {
        { ew,      ew + EW, EW, 0, 1 },   // author -> paper
        { ew + EW, ew,      EW, 1, 0 },   // paper  -> author
        { ea,      ea + EA, EA, 0, 2 },   // author -> inst
        { ea + EA, ea,      EA, 2, 0 },   // inst   -> author
        { et,      et + ET, ET, 1, 3 },   // paper  -> domain
        { et + ET, et,      ET, 3, 1 },   // domain -> paper
    };
    int nNodes[4]  = { NA, NP, NI, ND };
    int lastDir[4] = { 3, 5, 2, 4 };      // last dir index writing each dst type

    float* cnt[6];
    for (int i = 0; i < 6; ++i) cnt[i] = (float*)take((size_t)nNodes[dirs[i].dt] * 4);

    __bf16* linWt = (__bf16*)take((size_t)DIM * F * 2);            // 128 x F
    __bf16* Wcat1 = (__bf16*)take((size_t)6 * DIM * 2 * DIM * 2);  // 6 x (128 x 256)
    __bf16* Wcat2 = (__bf16*)take((size_t)6 * DIM * 2 * DIM * 2);
    (void)ws_size; (void)n_in; (void)out_size;

    const size_t gemmLds = (size_t)DIM * LDS_STRIDE * sizeof(__bf16);   // ~34 KB

    // ---- helpers ----
    auto zero = [&](float* p, long n) {
        k_zero<<<cdiv(n, 1024), 256, 0, stream>>>(p, n);
    };
    auto gemm = [&](const float* A0, int K0, const float* A1, int K1,
                    const __bf16* Wt, const float* b1, const float* b2,
                    float* C, int M, int flags) {
        k_gemm_bf16<<<cdiv(M, 128), 256, gemmLds, stream>>>(
            A0, K0, A1, K1, Wt, b1, b2, C, M, flags);
    };
    auto prep = [&](const float* W, __bf16* Wt, int K, int Ktot, int koff) {
        k_prep_w<<<cdiv((long)K * DIM, 256), 256, 0, stream>>>(W, Wt, K, Ktot, koff);
    };

    // ---- weight prep: stacked [Wl^T | Wr^T] panels per edge type ----
    prep(lin_W, linWt, F, F, 0);
    for (int i = 0; i < 6; ++i) {
        prep(Wl1 + (size_t)i * DIM * DIM, Wcat1 + (size_t)i * DIM * 2 * DIM, DIM, 2 * DIM, 0);
        prep(Wr1 + (size_t)i * DIM * DIM, Wcat1 + (size_t)i * DIM * 2 * DIM, DIM, 2 * DIM, DIM);
        prep(Wl2 + (size_t)i * DIM * DIM, Wcat2 + (size_t)i * DIM * 2 * DIM, DIM, 2 * DIM, 0);
        prep(Wr2 + (size_t)i * DIM * DIM, Wcat2 + (size_t)i * DIM * 2 * DIM, DIM, 2 * DIM, DIM);
    }

    // ---- node feature assembly ----
    k_gather<<<cdiv((long)NA * 32, 256), 256, 0, stream>>>(author_emb, author_nid, Xa, NA);
    k_gather<<<cdiv((long)NI * 32, 256), 256, 0, stream>>>(inst_emb,   inst_nid,   Xi, NI);
    k_gather<<<cdiv((long)ND * 32, 256), 256, 0, stream>>>(domain_emb, domain_nid, Xd, ND);
    k_gather<<<cdiv((long)NP * 32, 256), 256, 0, stream>>>(paper_emb,  paper_nid,  Xp, NP);
    // Xp += paper_x @ lin_W + lin_b
    gemm(paper_x, F, nullptr, 0, linWt, lin_b, nullptr, Xp, NP, 1);

    // ---- inverse segment counts (shared across both layers) ----
    for (int i = 0; i < 6; ++i) {
        int n = nNodes[dirs[i].dt];
        zero(cnt[i], n);
        k_count<<<cdiv(dirs[i].E, 256), 256, 0, stream>>>(dirs[i].d, cnt[i], dirs[i].E);
        k_inv<<<cdiv(n, 256), 256, 0, stream>>>(cnt[i], n);
    }

    // ---- hetero SAGE layer: per edge type one fused GEMM [agg | x_dt] @ [Wl^T|Wr^T] ----
    auto layer = [&](float* xin[4], float* xout[4], const __bf16* Wcat,
                     const float* bl, const float* br, bool doRelu) {
        bool started[4] = { false, false, false, false };
        for (int i = 0; i < 6; ++i) {
            int dt = dirs[i].dt;
            if (!xout[dt]) continue;
            int ndst = nNodes[dt];
            zero(AGG, (long)ndst * DIM);
            k_scatter<<<cdiv((long)dirs[i].E * 32, 256), 256, 0, stream>>>(
                xin[dirs[i].st], dirs[i].s, dirs[i].d, AGG, dirs[i].E);
            k_scale<<<cdiv((long)ndst * 32, 256), 256, 0, stream>>>(AGG, cnt[i], ndst);
            int flags = (started[dt] ? 1 : 0) |
                        ((doRelu && i == lastDir[dt]) ? 2 : 0);
            gemm(AGG, DIM, xin[dt], DIM, Wcat + (size_t)i * DIM * 2 * DIM,
                 bl + (size_t)i * DIM, br + (size_t)i * DIM, xout[dt], ndst, flags);
            started[dt] = true;
        }
    };

    float* X[4] = { Xa, Xp, Xi, Xd };
    float* H[4] = { Ha, Hp, Hi, Hd };
    layer(X, H, Wcat1, bl1, br1, true);            // layer 1 + fused relu

    // layer 2: only author and paper outputs are needed; reuse X buffers
    float* O[4] = { Xa, Xp, nullptr, nullptr };
    layer(H, O, Wcat2, bl2, br2, false);

    // ---- link prediction dot product ----
    k_dot<<<cdiv((long)EL * 32, 256), 256, 0, stream>>>(
        Xa, Xp, eli, eli + EL, (float*)d_out, EL);
}
